// InnocentAttention_34660386079325
// MI455X (gfx1250) — matrix operand order repaired and checked by reading, not compiled, not run on hardware
//
#include <hip/hip_runtime.h>
#include <hip/hip_bf16.h>

typedef __attribute__((ext_vector_type(16))) _Float16 v16h;
typedef __attribute__((ext_vector_type(8)))  _Float16 v8h;
typedef __attribute__((ext_vector_type(8)))  float    v8f;
typedef __attribute__((ext_vector_type(4)))  unsigned v4u;
typedef __attribute__((ext_vector_type(8)))  int      v8i;
typedef __attribute__((ext_vector_type(4)))  int      v4i;

#define S_LEN    2048
#define HEADD    128
#define NEG_INF  (-1e9f)
#define KV_TILES (S_LEN / 32)

#if defined(__has_builtin)
#if __has_builtin(__builtin_amdgcn_tensor_load_to_lds)
#define HAVE_TDM 1
#endif
#endif

// D = A x B + C, f16 inputs, f32 accumulate (CDNA5 wave32 WMMA)
static __device__ __forceinline__ v8f wmma_f16(v16h a, v16h b, v8f c) {
    return __builtin_amdgcn_wmma_f32_16x16x32_f16(
        false, a, false, b, (short)0, c, false, false);
}

#ifdef HAVE_TDM
// 1D TDM copy: nelem f16 elements, contiguous global -> contiguous LDS.
// D# per CDNA5 ISA ch.8: group0 = {count/type/addr}, group1 = {data_size, dims}.
static __device__ __forceinline__ void tdm_copy_f16(const void* gsrc,
                                                    unsigned lds_addr,
                                                    unsigned nelem) {
    unsigned long long ga = (unsigned long long)(uintptr_t)gsrc;
    v4u g0;
    g0[0] = 1u;                                           // count=1, user mode
    g0[1] = lds_addr;                                     // LDS byte address
    g0[2] = (unsigned)(ga & 0xFFFFFFFFu);                 // global_addr[31:0]
    g0[3] = (unsigned)((ga >> 32) & 0x01FFFFFFu)          // global_addr[56:32]
          | (2u << 30);                                   // type = 2 ("image")
    v8i g1 = (v8i)0;
    g1[0] = (int)(1u << 16);                              // data_size=1 -> 2 bytes
    g1[1] = (int)((nelem & 0xFFFFu) << 16);               // tensor_dim0[15:0]
    g1[2] = (int)(nelem >> 16);                           // tensor_dim0[31:16]
    g1[3] = (int)((nelem & 0xFFFFu) << 16);               // tile_dim0 (16-bit)
    v4i z = (v4i)0;
#if __clang_major__ >= 23
    v8i z8 = (v8i)0;
    __builtin_amdgcn_tensor_load_to_lds(g0, g1, z, z, z8, 0);
#else
    __builtin_amdgcn_tensor_load_to_lds(g0, g1, z, z, 0);
#endif
}
#endif

// ---- pass 1: K -> f16 row-major, V -> f16 transposed per 32-key tile ----
__global__ __launch_bounds__(256)
void InnocentAttention_prep_kv(const float* __restrict__ k,
                               const float* __restrict__ v,
                               _Float16* __restrict__ kh,     // [B*S, 128]
                               _Float16* __restrict__ vth,    // [B*S/32, 128, 32]
                               int total) {
    const int idx = (blockIdx.x * 256 + threadIdx.x) * 8;
    if (idx >= total) return;

    float4 a0 = ((const float4*)(k + idx))[0];
    float4 a1 = ((const float4*)(k + idx))[1];
    v8h h;
    h[0] = (_Float16)a0.x; h[1] = (_Float16)a0.y;
    h[2] = (_Float16)a0.z; h[3] = (_Float16)a0.w;
    h[4] = (_Float16)a1.x; h[5] = (_Float16)a1.y;
    h[6] = (_Float16)a1.z; h[7] = (_Float16)a1.w;
    *(v8h*)(kh + idx) = h;

    float4 b0 = ((const float4*)(v + idx))[0];
    float4 b1 = ((const float4*)(v + idx))[1];
    const int dcol = idx & (HEADD - 1);       // d .. d+7
    const int srow = idx / HEADD;             // b*S + s
    const int key  = srow & 31;
    const int tile = srow >> 5;               // global tile id = b*(S/32)+t
    _Float16* dst  = vth + ((size_t)tile * HEADD + dcol) * 32 + key;
    dst[0 * 32] = (_Float16)b0.x; dst[1 * 32] = (_Float16)b0.y;
    dst[2 * 32] = (_Float16)b0.z; dst[3 * 32] = (_Float16)b0.w;
    dst[4 * 32] = (_Float16)b1.x; dst[5 * 32] = (_Float16)b1.y;
    dst[6 * 32] = (_Float16)b1.z; dst[7 * 32] = (_Float16)b1.w;
}

// ---- pass 2: flash attention, TDM-staged, double-buffered ----
__global__ __launch_bounds__(128)
void InnocentAttention_fa_kernel(const float* __restrict__ q,
                                 const _Float16* __restrict__ kh,
                                 const _Float16* __restrict__ vth,
                                 const int*   __restrict__ event_length,
                                 float*       __restrict__ out)
{
    __shared__ __align__(32) _Float16 Ks[2][32][HEADD];   // 16 KB ping-pong
    __shared__ __align__(32) _Float16 Vt[2][HEADD][32];   // 16 KB ping-pong
    __shared__ __align__(32) _Float16 Ps[4][16][32];      // 4 KB per-wave P

    const int b    = blockIdx.y;
    const int qt   = blockIdx.x;
    const int tid  = threadIdx.x;
    const int wave = tid >> 5;
    const int lane = tid & 31;
    const int half = lane >> 4;
    const int l16  = lane & 15;
    const int kb   = half * 8;

    const int len = event_length[b];
    // (1/sqrt(128)) * log2(e): softmax done in exp2 domain -> bare v_exp_f32
    const float scale = 0.12751741030871432f;

    // Q A-fragments (16x32 per chunk), scale folded in
    const int    q_row_a = qt * 64 + wave * 16 + l16;
    const float* qrow    = q + ((size_t)b * S_LEN + q_row_a) * HEADD;
    v16h qa[4];
    #pragma unroll
    for (int c = 0; c < 4; ++c) {
        const float* p0 = qrow + 32 * c + 2 * kb;
        const float* p1 = qrow + 32 * c + 2 * kb + 8;
        float4 f0 = ((const float4*)p0)[0];
        float4 f1 = ((const float4*)p0)[1];
        float4 f2 = ((const float4*)p1)[0];
        float4 f3 = ((const float4*)p1)[1];
        v16h a;
        a[0]  = (_Float16)(f0.x * scale); a[1]  = (_Float16)(f0.y * scale);
        a[2]  = (_Float16)(f0.z * scale); a[3]  = (_Float16)(f0.w * scale);
        a[4]  = (_Float16)(f1.x * scale); a[5]  = (_Float16)(f1.y * scale);
        a[6]  = (_Float16)(f1.z * scale); a[7]  = (_Float16)(f1.w * scale);
        a[8]  = (_Float16)(f2.x * scale); a[9]  = (_Float16)(f2.y * scale);
        a[10] = (_Float16)(f2.z * scale); a[11] = (_Float16)(f2.w * scale);
        a[12] = (_Float16)(f3.x * scale); a[13] = (_Float16)(f3.y * scale);
        a[14] = (_Float16)(f3.z * scale); a[15] = (_Float16)(f3.w * scale);
        qa[c] = a;
    }

    v8f Oacc[8];
    #pragma unroll
    for (int db = 0; db < 8; ++db) Oacc[db] = 0;
    float m_r[8], l_r[8];
    #pragma unroll
    for (int r = 0; r < 8; ++r) { m_r[r] = -3.0e38f; l_r[r] = 0.0f; }

    const _Float16* khB  = kh  + (size_t)b * S_LEN * HEADD;
    const _Float16* vthB = vth + (size_t)b * KV_TILES * HEADD * 32;

#ifdef HAVE_TDM
    // prologue: DMA tile 0 into buffer 0 (wave 0 drives the TDM)
    if (wave == 0) {
        tdm_copy_f16(khB,  (unsigned)(uintptr_t)&Ks[0][0][0], 32 * HEADD);
        tdm_copy_f16(vthB, (unsigned)(uintptr_t)&Vt[0][0][0], 32 * HEADD);
    }
#endif

    for (int t = 0; t < KV_TILES; ++t) {
        const int cur = t & 1;
        const int kv  = t * 32;

#ifdef HAVE_TDM
        if (wave == 0) {
            if (t + 1 < KV_TILES) {
                const int nxt = cur ^ 1;
                tdm_copy_f16(khB + (size_t)(kv + 32) * HEADD,
                             (unsigned)(uintptr_t)&Ks[nxt][0][0], 32 * HEADD);
                tdm_copy_f16(vthB + (size_t)(t + 1) * HEADD * 32,
                             (unsigned)(uintptr_t)&Vt[nxt][0][0], 32 * HEADD);
                __builtin_amdgcn_s_wait_tensorcnt(2);   // tile t complete
            } else {
                __builtin_amdgcn_s_wait_tensorcnt(0);
            }
        }
#else
        {   // fallback: plain vector copy of the pre-converted f16 tiles
            _Float16*       dK = &Ks[cur][0][0] + tid * 32;
            _Float16*       dV = &Vt[cur][0][0] + tid * 32;
            const _Float16* sK = khB + (size_t)kv * HEADD + tid * 32;
            const _Float16* sV = vthB + (size_t)t * HEADD * 32 + tid * 32;
            ((v16h*)dK)[0] = ((const v16h*)sK)[0];
            ((v16h*)dK)[1] = ((const v16h*)sK)[1];
            ((v16h*)dV)[0] = ((const v16h*)sV)[0];
            ((v16h*)dV)[1] = ((const v16h*)sV)[1];
        }
#endif
        __syncthreads();

        // S = Q * K^T : two 16x16 score blocks
        v8f s0 = 0, s1 = 0;
        #pragma unroll
        for (int c = 0; c < 4; ++c) {
            v16h b0 = *(const v16h*)&Ks[cur][l16][32 * c + half * 16];
            s0 = wmma_f16(qa[c], b0, s0);
        }
        #pragma unroll
        for (int c = 0; c < 4; ++c) {
            v16h b1 = *(const v16h*)&Ks[cur][16 + l16][32 * c + half * 16];
            s1 = wmma_f16(qa[c], b1, s1);
        }

        // mask + online softmax (exp2 domain)
        const bool kvld0 = (kv + l16)      < len;
        const bool kvld1 = (kv + 16 + l16) < len;
        #pragma unroll
        for (int r = 0; r < 8; ++r) {
            const int  grow = qt * 64 + wave * 16 + half * 8 + r;
            const bool qvld = grow < len;
            float a0 = (qvld && kvld0) ? s0[r] : NEG_INF;
            float a1 = (qvld && kvld1) ? s1[r] : NEG_INF;

            float mx = fmaxf(a0, a1);
            mx = fmaxf(mx, __shfl_xor(mx, 1, 32));
            mx = fmaxf(mx, __shfl_xor(mx, 2, 32));
            mx = fmaxf(mx, __shfl_xor(mx, 4, 32));
            mx = fmaxf(mx, __shfl_xor(mx, 8, 32));

            float mnew = fmaxf(m_r[r], mx);
            float corr = exp2f(m_r[r] - mnew);
            float p0   = exp2f(a0 - mnew);
            float p1   = exp2f(a1 - mnew);
            float rs   = p0 + p1;
            rs += __shfl_xor(rs, 1, 32);
            rs += __shfl_xor(rs, 2, 32);
            rs += __shfl_xor(rs, 4, 32);
            rs += __shfl_xor(rs, 8, 32);

            l_r[r] = l_r[r] * corr + rs;
            m_r[r] = mnew;
            #pragma unroll
            for (int db = 0; db < 8; ++db) Oacc[db][r] *= corr;

            const int prow = half * 8 + r;
            Ps[wave][prow][l16]      = (_Float16)p0;
            Ps[wave][prow][16 + l16] = (_Float16)p1;
        }

        asm volatile("s_wait_dscnt 0" ::: "memory");  // wave-private P scratch

        v8h plo = *(const v8h*)&Ps[wave][l16][2 * kb];
        v8h phi = *(const v8h*)&Ps[wave][l16][2 * kb + 8];
        v16h pa = __builtin_shufflevector(plo, phi,
                     0,1,2,3,4,5,6,7,8,9,10,11,12,13,14,15);

        // O += P * V
        #pragma unroll
        for (int db = 0; db < 8; ++db) {
            v16h bv = *(const v16h*)&Vt[cur][16 * db + l16][half * 16];
            Oacc[db] = wmma_f16(pa, bv, Oacc[db]);
        }
        __syncthreads();
    }

    #pragma unroll
    for (int r = 0; r < 8; ++r) {
        const int grow = qt * 64 + wave * 16 + half * 8 + r;
        const float inv = 1.0f / l_r[r];
        float* orow = out + ((size_t)b * S_LEN + grow) * HEADD;
        #pragma unroll
        for (int db = 0; db < 8; ++db)
            orow[16 * db + l16] = Oacc[db][r] * inv;
    }
}

extern "C" void kernel_launch(void* const* d_in, const int* in_sizes, int n_in,
                              void* d_out, int out_size, void* d_ws, size_t ws_size,
                              hipStream_t stream) {
    const float* q   = (const float*)d_in[0];
    const float* k   = (const float*)d_in[1];
    const float* v   = (const float*)d_in[2];
    const int*   el  = (const int*)d_in[3];
    float*       out = (float*)d_out;

    const int B     = in_sizes[3];
    const int total = in_sizes[1];              // B*S*D elements of K
    // workspace: f16 K (row-major) + f16 V (tile-transposed); 2*total*2 bytes
    _Float16* kh  = (_Float16*)d_ws;
    _Float16* vth = kh + total;

    InnocentAttention_prep_kv<<<dim3((total / 8 + 255) / 256), dim3(256), 0, stream>>>(
        k, v, kh, vth, total);

    dim3 grid(S_LEN / 64, B);
    InnocentAttention_fa_kernel<<<grid, dim3(128), 0, stream>>>(q, kh, vth, el, out);
}
